// ResidualLogitAdapter_43989055046184
// MI455X (gfx1250) — matrix operand order repaired and checked by reading, not compiled, not run on hardware
//
#include <hip/hip_runtime.h>
#include <hip/hip_bf16.h>

// Problem constants (from reference)
#define NROWS 131072
#define GCOLS 512
#define FDIM  256
#define HDIM  128
#define KCOLS 64
#define KP    288   // GEMM1 K padded: 259 -> 9*32

typedef __attribute__((ext_vector_type(16))) __bf16 v16bf;
typedef __attribute__((ext_vector_type(8)))  float  v8f;

union Frag { v16bf v; unsigned short s[16]; uint4 q[2]; };

__device__ __forceinline__ unsigned short f2bf(float f) {
  unsigned int u = __float_as_uint(f);
  u += 0x7fffu + ((u >> 16) & 1u);   // round-to-nearest-even
  return (unsigned short)(u >> 16);
}

// Build bf16 transposed weight copies in workspace:
//   w1t[n][k] = W1[k][n] for k<259 else 0   (HDIM x KP)
//   w2t[n][h] = W2[h][n]                    (KCOLS x HDIM)
__global__ void prep_weights_kernel(const float* __restrict__ W1,
                                    const float* __restrict__ W2,
                                    unsigned short* __restrict__ w1t,
                                    unsigned short* __restrict__ w2t) {
  int i = blockIdx.x * blockDim.x + threadIdx.x;
  if (i < HDIM * KP) {
    int n = i / KP, k = i % KP;
    float v = (k < FDIM + 3) ? W1[(size_t)k * HDIM + n] : 0.0f;
    w1t[i] = f2bf(v);
  }
  int j = i - HDIM * KP;
  if (j >= 0 && j < KCOLS * HDIM) {
    int n = j / HDIM, h = j % HDIM;
    w2t[j] = f2bf(W2[(size_t)h * KCOLS + n]);
  }
}

__global__ __launch_bounds__(256)
void ResidualLogitAdapter_kernel(const float* __restrict__ z,
                                 const float* __restrict__ feats,
                                 const float* __restrict__ b1,
                                 const float* __restrict__ b2,
                                 const float* __restrict__ alphas,
                                 const int*   __restrict__ dom,
                                 const unsigned short* __restrict__ w1t,
                                 const unsigned short* __restrict__ w2t,
                                 float* __restrict__ out) {
  // Per-wave private LDS regions (8 waves/block); no cross-wave sharing,
  // so no workgroup barriers are required.
  __shared__ __align__(16) unsigned short clds[8][16 * 32];   // conf stats, zero-padded K-panel
  __shared__ __align__(16) unsigned short Hlds[8][16 * HDIM]; // hidden acts, bf16
  __shared__ float aRowS[8][16];
  __shared__ int   domS[8][16];

  const int lane = threadIdx.x & 31;
  const int wv   = threadIdx.x >> 5;
  const int r0   = blockIdx.x * 128 + wv * 16;   // this wave's 16 rows

  // ---------- Phase 0: stream-copy z -> out (memory-bound bulk) ----------
  for (int i = lane; i < 16 * (GCOLS / 4); i += 32) {
    int r  = i >> 7;
    int c4 = i & 127;
    size_t off = (size_t)(r0 + r) * GCOLS + (size_t)c4 * 4;
    float4 vv = *(const float4*)(z + off);
    *(float4*)(out + off) = vv;
  }

  // ---------- Phase 1: per-row confidence stats ----------
  for (int i = lane; i < 16 * 32; i += 32) clds[wv][i] = 0;

  for (int r = 0; r < 16; ++r) {
    int row = r0 + r;
    int d   = dom[row];
    const float* zr = z + (size_t)row * GCOLS + (size_t)d * KCOLS;
    float v0 = zr[lane], v1 = zr[lane + 32];
    // top-2 butterfly reduction over 64 values
    float m1 = fmaxf(v0, v1), m2 = fminf(v0, v1);
#pragma unroll
    for (int off = 16; off; off >>= 1) {
      float o1 = __shfl_xor(m1, off, 32);
      float o2 = __shfl_xor(m2, off, 32);
      float n1 = fmaxf(m1, o1);
      float n2 = fmaxf(fminf(m1, o1), fmaxf(m2, o2));
      m1 = n1; m2 = n2;
    }
    float e0 = __expf(v0 - m1), e1 = __expf(v1 - m1);
    float s  = e0 + e1;
    float ws = e0 * (v0 - m1) + e1 * (v1 - m1);
#pragma unroll
    for (int off = 16; off; off >>= 1) {
      s  += __shfl_xor(s,  off, 32);
      ws += __shfl_xor(ws, off, 32);
    }
    if (lane == 0) {
      float inv = 1.0f / s;
      clds[wv][r * 32 + 0] = f2bf(inv);                            // p_max = e^0 / S
      clds[wv][r * 32 + 1] = f2bf(__logf(s) - ws * inv);           // entropy
      clds[wv][r * 32 + 2] = f2bf((1.0f - __expf(m2 - m1)) * inv); // top-2 margin
      aRowS[wv][r] = alphas[d];
      domS[wv][r]  = d;
    }
  }

  // ---------- GEMM1: h = relu([feats|c|0] @ W1 + b1), bf16 WMMA, f32 acc ----------
  // 16-bit A (16x32) layout: lane -> M = lane&15; K chunks [koff,koff+8) and
  // [16+koff,16+koff+8) with koff = lane<16 ? 0 : 8.
  const int rA   = lane & 15;
  const int koff = (lane < 16) ? 0 : 8;
  const int bko  = (lane < 16) ? 0 : 8;   // B (32x16): lane = col N, K chunks [bko,bko+8) and [16+bko,16+bko+8)
  const int mOff = (lane < 16) ? 0 : 8;   // C/D: vgpr j -> M = j + mOff

  Frag afr[9];
#pragma unroll
  for (int k = 0; k < 8; ++k) {
    const float* fr = feats + (size_t)(r0 + rA) * FDIM + k * 32;
    float4 f0 = *(const float4*)(fr + koff);
    float4 f1 = *(const float4*)(fr + koff + 4);
    float4 f2 = *(const float4*)(fr + 16 + koff);
    float4 f3 = *(const float4*)(fr + 16 + koff + 4);
    afr[k].s[0]  = f2bf(f0.x); afr[k].s[1]  = f2bf(f0.y);
    afr[k].s[2]  = f2bf(f0.z); afr[k].s[3]  = f2bf(f0.w);
    afr[k].s[4]  = f2bf(f1.x); afr[k].s[5]  = f2bf(f1.y);
    afr[k].s[6]  = f2bf(f1.z); afr[k].s[7]  = f2bf(f1.w);
    afr[k].s[8]  = f2bf(f2.x); afr[k].s[9]  = f2bf(f2.y);
    afr[k].s[10] = f2bf(f2.z); afr[k].s[11] = f2bf(f2.w);
    afr[k].s[12] = f2bf(f3.x); afr[k].s[13] = f2bf(f3.y);
    afr[k].s[14] = f2bf(f3.z); afr[k].s[15] = f2bf(f3.w);
  }
  { // K-step 8: [c0 c1 c2 0...0] panel from LDS
    const unsigned short* cp = &clds[wv][rA * 32];
    afr[8].q[0] = *(const uint4*)(cp + koff);
    afr[8].q[1] = *(const uint4*)(cp + 16 + koff);
  }

#pragma unroll
  for (int nt = 0; nt < 8; ++nt) {
    float bias = b1[nt * 16 + (lane & 15)];
    v8f acc;
#pragma unroll
    for (int j = 0; j < 8; ++j) acc[j] = bias;
#pragma unroll
    for (int k = 0; k < 9; ++k) {
      Frag bfr;
      const unsigned short* bp =
          w1t + (size_t)(nt * 16 + (lane & 15)) * KP + k * 32 + bko;
      bfr.q[0] = *(const uint4*)(bp);
      bfr.q[1] = *(const uint4*)(bp + 16);
      acc = __builtin_amdgcn_wmma_f32_16x16x32_bf16(
          false, afr[k].v, false, bfr.v, (short)0, acc, false, false);
    }
#pragma unroll
    for (int j = 0; j < 8; ++j) {
      float h = fmaxf(acc[j], 0.0f);  // relu
      Hlds[wv][(j + mOff) * HDIM + nt * 16 + (lane & 15)] = f2bf(h);
    }
  }

  // ---------- GEMM2: dz = (h @ W2 + b2) * alpha; RMW-add into owned cols ----------
  Frag hfr[4];
#pragma unroll
  for (int k = 0; k < 4; ++k) {
    const unsigned short* hp = &Hlds[wv][rA * HDIM + k * 32];
    hfr[k].q[0] = *(const uint4*)(hp + koff);
    hfr[k].q[1] = *(const uint4*)(hp + 16 + koff);
  }

#pragma unroll
  for (int nt = 0; nt < 4; ++nt) {
    float bias = b2[nt * 16 + (lane & 15)];
    v8f acc;
#pragma unroll
    for (int j = 0; j < 8; ++j) acc[j] = bias;
#pragma unroll
    for (int k = 0; k < 4; ++k) {
      Frag bfr;
      const unsigned short* bp =
          w2t + (size_t)(nt * 16 + (lane & 15)) * HDIM + k * 32 + bko;
      bfr.q[0] = *(const uint4*)(bp);
      bfr.q[1] = *(const uint4*)(bp + 16);
      acc = __builtin_amdgcn_wmma_f32_16x16x32_bf16(
          false, hfr[k].v, false, bfr.v, (short)0, acc, false, false);
    }
#pragma unroll
    for (int j = 0; j < 8; ++j) {
      int   M = j + mOff;
      float a = aRowS[wv][M];
      int   d = domS[wv][M];
      size_t o = (size_t)(r0 + M) * GCOLS + (size_t)d * KCOLS
               + (size_t)(nt * 16 + (lane & 15));
      // same-wave store->load to same address is kept in order by HW
      out[o] += acc[j] * a;
    }
  }
}

extern "C" void kernel_launch(void* const* d_in, const int* in_sizes, int n_in,
                              void* d_out, int out_size, void* d_ws, size_t ws_size,
                              hipStream_t stream) {
  const float* z      = (const float*)d_in[0];
  const float* feats  = (const float*)d_in[1];
  const float* W1     = (const float*)d_in[2];
  const float* b1     = (const float*)d_in[3];
  const float* W2     = (const float*)d_in[4];
  const float* b2     = (const float*)d_in[5];
  const float* alphas = (const float*)d_in[6];
  const int*   dom    = (const int*)d_in[7];

  unsigned short* w1t = (unsigned short*)d_ws;            // HDIM*KP bf16
  unsigned short* w2t = w1t + (size_t)HDIM * KP;          // KCOLS*HDIM bf16

  int prepThreads = HDIM * KP + KCOLS * HDIM;
  prep_weights_kernel<<<(prepThreads + 255) / 256, 256, 0, stream>>>(W1, W2, w1t, w2t);

  ResidualLogitAdapter_kernel<<<NROWS / 128, 256, 0, stream>>>(
      z, feats, b1, b2, alphas, dom, w1t, w2t, (float*)d_out);
}
